// ASTPathsEncoder_9895604650635
// MI455X (gfx1250) — matrix operand order repaired and checked by reading, not compiled, not run on hardware
//
#include <hip/hip_runtime.h>
#include <hip/hip_bf16.h>
#include <stdint.h>

// ---------------------------------------------------------------------------
// ASTPathsEncoder for MI455X (gfx1250, wave32, WMMA)
//
// Pipeline:
//  1) zero new_nodes
//  2) convert ast_nodes f32->bf16 (row-major pairs)
//  3) pack gru_Wx / gru_Wh f32->bf16 in WMMA B-fragment-major layout
//  4) combo_table[256]  = concat(emb[a],emb[b]) @ proj_W + proj_b   (tiny)
//  5) combo_xg[256,768] = combo_table @ Wx + bx                     (tiny)
//  6) ast_xg[50000,768] = ast_nodes @ Wx + bx   -- WMMA bf16 GEMM, L2-resident
//  7) GRU recurrence: 64 steps, per block 32 paths (2 M-tiles so each Wh
//     B-fragment feeds 2 WMMAs); hg = h @ Wh via v_wmma_f32_16x16x32_bf16;
//     gates in registers; outputs + fused segment-sum (global_atomic_add_f32)
// ---------------------------------------------------------------------------

#define NN   50000
#define PP   8192
#define LL   32
#define DD   256
#define TT   64
#define K3   768          // 3*D
#define NT48 48           // 768/16 N-tiles
#define KT8  8            // 256/32 K-steps
#define MB   2            // M-tiles (16 paths each) per recurrence block

typedef __attribute__((ext_vector_type(16))) __bf16 v16bf;
typedef __attribute__((ext_vector_type(8)))  float  v8f;

union Frag32 { uint4 q[2]; v16bf v; };

__device__ __forceinline__ unsigned short f32_to_bf16(float f) {
  unsigned int u = __float_as_uint(f);
  unsigned int r = u + 0x7FFFu + ((u >> 16) & 1u);   // round-to-nearest-even
  return (unsigned short)(r >> 16);
}

// A fragment (16x32 bf16) from row-major bf16 stored as dword pairs.
// lane m (m<16): row m, K in {0..7, 16..23}; lane 16+m: row m, K {8..15, 24..31}.
__device__ __forceinline__ v16bf load_a_bf16(const unsigned int* base_dw, int ld_dw,
                                             int kt, int lane) {
  int row = lane & 15;
  int off = (lane < 16) ? 0 : 4;                // dwords (8 bf16 = 4 dwords)
  const unsigned int* p = base_dw + row * ld_dw + kt * 16 + off;
  Frag32 f;
  f.q[0] = *reinterpret_cast<const uint4*>(p);
  f.q[1] = *reinterpret_cast<const uint4*>(p + 8);
  return f.v;
}

// B fragment (32x16 bf16) from pre-swizzled weights: contiguous 8 dwords/lane.
__device__ __forceinline__ v16bf load_b_packed(const unsigned int* packed,
                                               int kt, int nt, int lane) {
  const unsigned int* p = packed + (((kt * NT48 + nt) * 32 + lane) << 3);
  Frag32 f;
  f.q[0] = *reinterpret_cast<const uint4*>(p);
  f.q[1] = *reinterpret_cast<const uint4*>(p + 4);
  return f.v;
}

// ---------------------------------------------------------------------------
__global__ void zero_f32(float* __restrict__ p, size_t n) {
  size_t i = (size_t)blockIdx.x * blockDim.x + threadIdx.x;
  size_t stride = (size_t)gridDim.x * blockDim.x;
  for (; i < n; i += stride) p[i] = 0.0f;
}

__global__ void cvt_ast_bf16(const float* __restrict__ ast, unsigned int* __restrict__ astW) {
  int i = blockIdx.x * blockDim.x + threadIdx.x;     // dword index, N*128 total
  if (i >= NN * (DD / 2)) return;
  float a = ast[2 * i], b = ast[2 * i + 1];
  astW[i] = (unsigned int)f32_to_bf16(a) | ((unsigned int)f32_to_bf16(b) << 16);
}

// Pack W[256,768] f32 -> bf16 in B-fragment-major order:
// packed[((kt*48+nt)*32+lane)*8 + v] = {W[kb+2(v&3)+16(v>>2)][col], W[...+1][col]},
// col = nt*16 + (lane&15), kb = kt*32 + (lane<16?0:8).
__global__ void pack_weight_bf16(const float* __restrict__ W, unsigned int* __restrict__ packed) {
  int t = blockIdx.x * blockDim.x + threadIdx.x;
  if (t >= KT8 * NT48 * 32 * 8) return;
  int v    = t & 7;
  int lane = (t >> 3) & 31;
  int r2   = t >> 8;
  int nt   = r2 % NT48;
  int kt   = r2 / NT48;
  int col  = nt * 16 + (lane & 15);
  int k0   = kt * 32 + ((lane >> 4) << 3) + 2 * (v & 3) + ((v >> 2) << 4);
  unsigned int lo = f32_to_bf16(W[k0 * K3 + col]);
  unsigned int hi = f32_to_bf16(W[(k0 + 1) * K3 + col]);
  packed[t] = lo | (hi << 16);
}

// combo_table[c][d] = concat(emb[c>>4], emb[c&15]) @ proj_W + proj_b
__global__ void combo_table_kernel(const float* __restrict__ emb, const float* __restrict__ projW,
                                   const float* __restrict__ projb, float* __restrict__ combo) {
  int c = blockIdx.x, d = threadIdx.x;
  int a = c >> 4, b = c & 15;
  float acc = projb[d];
  #pragma unroll 4
  for (int j = 0; j < DD; ++j) acc += emb[a * DD + j] * projW[j * DD + d];
  #pragma unroll 4
  for (int j = 0; j < DD; ++j) acc += emb[b * DD + j] * projW[(DD + j) * DD + d];
  combo[c * DD + d] = acc;
}

// combo_xg[c][k] = combo_table[c] @ Wx + bx
__global__ void combo_xg_kernel(const float* __restrict__ combo, const float* __restrict__ Wx,
                                const float* __restrict__ bx, float* __restrict__ cxg) {
  int c = blockIdx.x;
  int k = blockIdx.y * 256 + threadIdx.x;
  float acc = bx[k];
  #pragma unroll 4
  for (int dd = 0; dd < DD; ++dd) acc += combo[c * DD + dd] * Wx[dd * K3 + k];
  cxg[c * K3 + k] = acc;
}

// Wave w's 6 N-tiles: matching r/z/n columns d in [32w, 32w+32)
__device__ __forceinline__ void nt_list(int w, int* ntl) {
  ntl[0] = 2 * w;      ntl[1] = 2 * w + 1;
  ntl[2] = 16 + 2 * w; ntl[3] = 16 + 2 * w + 1;
  ntl[4] = 32 + 2 * w; ntl[5] = 32 + 2 * w + 1;
}

// ast_xg[50000,768] = ast_nodes(bf16) @ WxPacked + bx  (WMMA bf16 GEMM)
__global__ __launch_bounds__(256) void astxg_gemm(const unsigned int* __restrict__ astW,
                                                  const unsigned int* __restrict__ WxP,
                                                  const float* __restrict__ bx,
                                                  float* __restrict__ ast_xg) {
  int tid = threadIdx.x, wid = tid >> 5, lane = tid & 31;
  int mt = blockIdx.x;                                   // 0..3124 (50000/16)
  const unsigned int* abase = astW + (size_t)mt * 16 * (DD / 2);
  int ntl[6]; nt_list(wid, ntl);
  v8f acc[6] = {};
  #pragma unroll
  for (int kt = 0; kt < KT8; ++kt) {
    v16bf a = load_a_bf16(abase, DD / 2, kt, lane);
    if (kt < KT8 - 1)
      __builtin_prefetch(WxP + ((((kt + 1) * NT48 + ntl[0]) * 32 + lane) << 3), 0, 3);
    #pragma unroll
    for (int j = 0; j < 6; ++j) {
      v16bf b = load_b_packed(WxP, kt, ntl[j], lane);
      acc[j] = __builtin_amdgcn_wmma_f32_16x16x32_bf16(false, a, false, b,
                                                       (short)0, acc[j], false, false);
    }
  }
  int col = lane & 15, rowOff = (lane < 16) ? 0 : 8;
  #pragma unroll
  for (int j = 0; j < 6; ++j) {
    int nt = ntl[j];
    float bias = bx[nt * 16 + col];
    #pragma unroll
    for (int r = 0; r < 8; ++r) {
      int row = mt * 16 + rowOff + r;
      ast_xg[(size_t)row * K3 + nt * 16 + col] = acc[j][r] + bias;
    }
  }
}

// GRU recurrence: 1 block = 32 paths (2 M-tiles), 64 timesteps.
// Each Wh B-fragment load feeds MB=2 WMMAs (halves L2 weight traffic).
__global__ __launch_bounds__(256) void gru_recur(const unsigned int* __restrict__ WhP,
                                                 const float* __restrict__ ast_xg,
                                                 const float* __restrict__ combo_xg,
                                                 const int* __restrict__ path_idx,
                                                 const int* __restrict__ seq_len,
                                                 const int* __restrict__ child_place,
                                                 const int* __restrict__ vert_dir,
                                                 const float* __restrict__ bh,
                                                 float* __restrict__ new_nodes,
                                                 float* __restrict__ out_nodes,
                                                 float* __restrict__ out_orient) {
  __shared__ float          h_s[MB * 16 * DD];     // fp32 master hidden state (32 KB)
  __shared__ unsigned short hb_s[MB * 16 * DD];    // bf16 copy, WMMA A operand (16 KB)
  __shared__ const float*   rowptr[MB * 16];       // per-path xg row for step t
  __shared__ int            nidx_s[MB * 16];       // node index (even steps)
  __shared__ int            slen_s[MB * 16];

  int tid = threadIdx.x, wid = tid >> 5, lane = tid & 31;
  int pbase = blockIdx.x * (MB * 16);

  for (int i = tid; i < MB * 16 * DD; i += 256) { h_s[i] = 0.0f; hb_s[i] = 0; }
  if (tid < MB * 16) slen_s[tid] = seq_len[pbase + tid];

  int col = lane & 15;
  int rowOff = (lane < 16) ? 0 : 8;
  int ntl[6]; nt_list(wid, ntl);

  // hoist bh for this wave's 32 columns
  float bhv[6];
  #pragma unroll
  for (int j = 0; j < 2; ++j) {
    int d = 32 * wid + 16 * j + col;
    bhv[j]     = bh[d];
    bhv[2 + j] = bh[DD + d];
    bhv[4 + j] = bh[2 * DD + d];
  }

  const unsigned int* hb_dw = reinterpret_cast<const unsigned int*>(hb_s);

  for (int t = 0; t < TT; ++t) {
    __syncthreads();                       // hb_s from prev step visible; prev gates done
    if (tid < MB * 16) {                   // per-path xg row pointer for this step
      int p = pbase + tid, l = t >> 1;
      if ((t & 1) == 0) {
        int nid = path_idx[p * LL + l];
        nidx_s[tid] = nid;
        rowptr[tid] = ast_xg + (size_t)nid * K3;
      } else {
        int c = child_place[p * LL + l] * 16 + vert_dir[p * LL + l];
        rowptr[tid] = combo_xg + (size_t)c * K3;
      }
    }

    // hg = h(32x256 bf16) @ Wh(256x768 bf16): per wave 2 M-tiles x 6 N-tiles
    v8f acc[MB][6] = {};
    #pragma unroll
    for (int kt = 0; kt < KT8; ++kt) {
      v16bf a0 = load_a_bf16(hb_dw, DD / 2, kt, lane);
      v16bf a1 = load_a_bf16(hb_dw + 16 * (DD / 2), DD / 2, kt, lane);
      if (kt < KT8 - 1)
        __builtin_prefetch(WhP + ((((kt + 1) * NT48 + ntl[0]) * 32 + lane) << 3), 0, 3);
      #pragma unroll
      for (int j = 0; j < 6; ++j) {
        v16bf b = load_b_packed(WhP, kt, ntl[j], lane);
        acc[0][j] = __builtin_amdgcn_wmma_f32_16x16x32_bf16(false, a0, false, b,
                                                            (short)0, acc[0][j], false, false);
        acc[1][j] = __builtin_amdgcn_wmma_f32_16x16x32_bf16(false, a1, false, b,
                                                            (short)0, acc[1][j], false, false);
      }
    }
    __syncthreads();                       // all hb_s reads complete; rowptr visible

    int l = t >> 1;
    bool isNode = (t & 1) == 0;
    float* outb = isNode ? out_nodes : out_orient;

    #pragma unroll
    for (int mb = 0; mb < MB; ++mb) {
      #pragma unroll
      for (int j = 0; j < 2; ++j) {
        int d = 32 * wid + 16 * j + col;
        #pragma unroll
        for (int r = 0; r < 8; ++r) {
          int row = mb * 16 + rowOff + r;
          const float* rp = rowptr[row];
          float xr = rp[d], xz = rp[DD + d], xn = rp[2 * DD + d];
          float hr = acc[mb][j][r]     + bhv[j];
          float hz = acc[mb][2 + j][r] + bhv[2 + j];
          float hn = acc[mb][4 + j][r] + bhv[4 + j];
          float rg = 1.0f / (1.0f + __expf(-(xr + hr)));
          float zg = 1.0f / (1.0f + __expf(-(xz + hz)));
          float ng = tanhf(xn + rg * hn);
          float ho = h_s[row * DD + d];
          bool valid = t < 2 * slen_s[row];
          float hnew = valid ? ((1.0f - zg) * ng + zg * ho) : ho;
          h_s[row * DD + d]  = hnew;
          hb_s[row * DD + d] = f32_to_bf16(hnew);
          int p = pbase + row;
          outb[((size_t)p * LL + l) * DD + d] = hnew;
          if (isNode && l < slen_s[row])     // fused segment_sum
            atomicAdd(&new_nodes[(size_t)nidx_s[row] * DD + d], hnew);
        }
      }
    }
  }
}

// ---------------------------------------------------------------------------
extern "C" void kernel_launch(void* const* d_in, const int* in_sizes, int n_in,
                              void* d_out, int out_size, void* d_ws, size_t ws_size,
                              hipStream_t stream) {
  const float* ast   = (const float*)d_in[0];
  const int*   idx   = (const int*)d_in[1];
  const int*   slen  = (const int*)d_in[2];
  const int*   cp    = (const int*)d_in[3];
  const int*   vd    = (const int*)d_in[4];
  const float* emb   = (const float*)d_in[5];
  const float* projW = (const float*)d_in[6];
  const float* projb = (const float*)d_in[7];
  const float* Wx    = (const float*)d_in[8];
  const float* Wh    = (const float*)d_in[9];
  const float* bx    = (const float*)d_in[10];
  const float* bh    = (const float*)d_in[11];

  float* out_new    = (float*)d_out;                              // (N, D)
  float* out_nodes  = out_new + (size_t)NN * DD;                  // (P, L, D)
  float* out_orient = out_nodes + (size_t)PP * LL * DD;           // (P, L, D)

  // workspace carve-out (~181 MB total)
  char* ws = (char*)d_ws;
  auto carve = [&](size_t bytes) { char* p = ws; ws += (bytes + 255) & ~(size_t)255; return p; };
  unsigned int* astW    = (unsigned int*)carve((size_t)NN * (DD / 2) * 4);
  unsigned int* WxP     = (unsigned int*)carve((size_t)KT8 * NT48 * 32 * 8 * 4);
  unsigned int* WhP     = (unsigned int*)carve((size_t)KT8 * NT48 * 32 * 8 * 4);
  float*        combo_t = (float*)carve((size_t)256 * DD * 4);
  float*        comboxg = (float*)carve((size_t)256 * K3 * 4);
  float*        ast_xg  = (float*)carve((size_t)NN * K3 * 4);

  zero_f32<<<8192, 256, 0, stream>>>(out_new, (size_t)NN * DD);
  cvt_ast_bf16<<<(NN * (DD / 2) + 255) / 256, 256, 0, stream>>>(ast, astW);
  pack_weight_bf16<<<(KT8 * NT48 * 32 * 8) / 256, 256, 0, stream>>>(Wx, WxP);
  pack_weight_bf16<<<(KT8 * NT48 * 32 * 8) / 256, 256, 0, stream>>>(Wh, WhP);
  combo_table_kernel<<<256, 256, 0, stream>>>(emb, projW, projb, combo_t);
  combo_xg_kernel<<<dim3(256, 3), 256, 0, stream>>>(combo_t, Wx, bx, comboxg);
  astxg_gemm<<<NN / 16, 256, 0, stream>>>(astW, WxP, bx, ast_xg);
  gru_recur<<<PP / (MB * 16), 256, 0, stream>>>(WhP, ast_xg, comboxg, idx, slen, cp, vd,
                                                bh, out_new, out_nodes, out_orient);
}
